// CustomPedestrianDetector_74698071212177
// MI455X (gfx1250) — compile-verified, matrix operand order checked by reading, not run on hardware
//
#include <hip/hip_runtime.h>
#include <hip/hip_bf16.h>
#include <math.h>

// ---------------------------------------------------------------------------
// CDNA5 (gfx1250) pedestrian-detector pipeline.
//  * All matrix stages: implicit GEMM on V_WMMA_F32_16X16X32_BF16,
//    128x64x32 LDS-tiled (8 waves; each wave: 1 A-frag, 4 B-frags, 4 WMMA/K-step).
//  * Weight/W tiles DMA'd into LDS by the Tensor Data Mover
//    (tensor_load_to_lds + s_wait_tensorcnt), with D# LDS padding giving a
//    bank-conflict-free [row][17-dword] layout. Weights pre-converted to a
//    bf16 arena once per call (2.6 MB, trivial at 23.3 TB/s).
//  * Activations: f32 in HBM for convs (im2col gather + convert while staging);
//    pooled/hidden FC tensors stored bf16.
// ---------------------------------------------------------------------------

typedef __attribute__((ext_vector_type(16))) __bf16        v16bf;
typedef __attribute__((ext_vector_type(8)))  float          v8f;
typedef __attribute__((ext_vector_type(4)))  unsigned int   u32x4;
typedef __attribute__((ext_vector_type(8)))  int            i32x8;
typedef __attribute__((ext_vector_type(4)))  int            i32x4;

union FragBF {
    v16bf v;
    unsigned int u[8];
};

static __device__ __forceinline__ unsigned short f2bf(float f) {
    union { float f; unsigned int u; } cv; cv.f = f;
    const unsigned int u = cv.u;
    const unsigned int rounding = 0x7FFFu + ((u >> 16) & 1u);
    return (unsigned short)((u + rounding) >> 16);
}
static __device__ __forceinline__ unsigned int packbf(float lo, float hi) {
    return (unsigned int)f2bf(lo) | ((unsigned int)f2bf(hi) << 16);
}

#define BM 128
#define BN 64
#define BK 32
#define LDW (BK / 2 + 1)   // 17 dwords per row (16 data + 1 pad)

// ---------------------------------------------------------------------------
// TDM: DMA a [tile_h rows x tile_w bf16] 2D tile (row stride row_stride elems)
// into LDS at lds_addr, padding +1 DWORD after every 16 DWORDs (-> [row][17]).
// Rows/cols beyond (tensor_h, tensor_w) are zero-filled by the TDM OOB rule.
// ---------------------------------------------------------------------------
static __device__ __forceinline__ void tdm_load_2d_bf16(
    unsigned int lds_addr, const void* gaddr,
    unsigned int tile_w, unsigned int tile_h,
    unsigned long long row_stride,
    unsigned int tensor_w, unsigned int tensor_h)
{
#if __has_builtin(__builtin_amdgcn_tensor_load_to_lds)
    const unsigned long long ga = (unsigned long long)gaddr;
    u32x4 g0;
    g0[0] = 1u;                                              // count=1, user D#
    g0[1] = lds_addr;                                        // LDS byte address
    g0[2] = (unsigned int)(ga & 0xFFFFFFFFu);                // global_addr[31:0]
    g0[3] = (unsigned int)((ga >> 32) & 0x01FFFFFFu)         // global_addr[56:32]
            | (2u << 30);                                    // type = 2 ("image")
    i32x8 g1;
    // data_size=1 (2B), pad_enable=1, pad_interval=3 (16 DW), pad_amount=0 (1 DW)
    g1[0] = (int)((1u << 16) | (1u << 20) | (3u << 22));
    g1[1] = (int)((tensor_w & 0xFFFFu) << 16);               // tensor_dim0[15:0]
    g1[2] = (int)((tensor_w >> 16) | ((tensor_h & 0xFFFFu) << 16));
    g1[3] = (int)((tensor_h >> 16) | (tile_w << 16));        // tile_dim0
    g1[4] = (int)(tile_h & 0xFFFFu);                         // tile_dim1 (dim2=0)
    g1[5] = (int)(row_stride & 0xFFFFFFFFull);               // dim0_stride lo
    g1[6] = (int)((row_stride >> 32) & 0xFFFFull);           // dim0_stride hi
    g1[7] = 0;
    const i32x4 z4 = {0, 0, 0, 0};
#if defined(__clang_major__) && (__clang_major__ >= 23)
    const i32x8 z8 = {0, 0, 0, 0, 0, 0, 0, 0};
    __builtin_amdgcn_tensor_load_to_lds(g0, g1, z4, z4, z8, 0);
#else
    __builtin_amdgcn_tensor_load_to_lds(g0, g1, z4, z4, 0);
#endif
#endif
}

// f32 -> bf16 conversion (weight arena prep)
__global__ void f2bf_kernel(const float* __restrict__ in,
                            unsigned short* __restrict__ out, int n)
{
    const int i = blockIdx.x * blockDim.x + threadIdx.x;
    if (i < n) out[i] = f2bf(in[i]);
}

// ---------------------------------------------------------------------------
// Implicit-GEMM conv: y[NCHW] = act( conv(x, wbf) * scale + shift )
// mode: 0 = bias, 1 = bias+relu, 2 = batchnorm(g,b,m,v)+relu
// Block: 256 threads = 8 waves; wave mt = 16-row M subtile; each wave covers
// all four 16-col N subtiles (4 WMMA / K-step, A-fragment reused).
// Weight tile (64 x 32 bf16) arrives via the Tensor Data Mover.
// ---------------------------------------------------------------------------
__global__ __launch_bounds__(256) void conv_wmma_kernel(
    const float* __restrict__ x, const unsigned short* __restrict__ wbf,
    const float* __restrict__ p0, const float* __restrict__ p1,
    const float* __restrict__ p2, const float* __restrict__ p3,
    float* __restrict__ y,
    int Bn, int Cin, int H, int W, int Cout, int OH, int OW,
    int stride, int pad, int ks, int mode)
{
    __shared__ unsigned int As[BM][LDW];
    __shared__ unsigned int Bs[BN][LDW];

    const int tid  = threadIdx.x;
    const int lane = tid & 31;
    const int wv   = tid >> 5;          // wave id == M subtile
    const int l16  = lane & 15;
    const int h    = lane >> 4;

    const int hw  = OH * OW;
    const int M   = Bn * hw;
    const int K   = Cin * ks * ks;
    const int ks2 = ks * ks;
    const int m0  = blockIdx.x * BM;
    const int n0  = blockIdx.y * BN;

    // hoist im2col row decomposition for this thread's 8 staged A pairs
    int aib[8], aiy0[8], aix0[8];
#pragma unroll
    for (int j = 0; j < 8; ++j) {
        const int uidx = tid + 256 * j;      // 0..2047
        const int row  = uidx >> 4;          // 0..127
        int mA = m0 + row; if (mA >= M) mA = M - 1;
        const int ab   = mA / hw;
        const int arem = mA - ab * hw;
        const int aoy  = arem / OW;
        const int aox  = arem - aoy * OW;
        aib[j]  = ab;
        aiy0[j] = aoy * stride - pad;
        aix0[j] = aox * stride - pad;
    }

    v8f acc[4] = {v8f{}, v8f{}, v8f{}, v8f{}};

    for (int kb = 0; kb < K; kb += BK) {
        __syncthreads();   // previous iteration's LDS reads complete
        // ---- TDM: weight tile [64 x 32] bf16 -> Bs (padded rows) ----
        if (wv == 0) {
            tdm_load_2d_bf16((unsigned int)(size_t)&Bs[0][0],
                             wbf + (size_t)n0 * K + kb,
                             BK, BN, (unsigned long long)K,
                             (unsigned int)(K - kb), (unsigned int)(Cout - n0));
        }
        // ---- stage A (im2col gather, f32->bf16) ----
#pragma unroll
        for (int j = 0; j < 8; ++j) {
            const int uidx = tid + 256 * j;
            const int row  = uidx >> 4;
            const int ucol = uidx & 15;
            const int k0   = kb + ucol * 2;
            float a0 = 0.0f, a1 = 0.0f;
#pragma unroll
            for (int e = 0; e < 2; ++e) {
                const int k = k0 + e;
                float av = 0.0f;
                if (k < K) {
                    const int ci = k / ks2;
                    const int r2 = k - ci * ks2;
                    const int ky = r2 / ks;
                    const int kx = r2 - ky * ks;
                    const int iy = aiy0[j] + ky;
                    const int ix = aix0[j] + kx;
                    if (iy >= 0 && iy < H && ix >= 0 && ix < W)
                        av = x[(((size_t)aib[j] * Cin + ci) * H + iy) * W + ix];
                }
                if (e == 0) a0 = av; else a1 = av;
            }
            As[row][ucol] = packbf(a0, a1);
        }
#if __has_builtin(__builtin_amdgcn_s_wait_tensorcnt)
        if (wv == 0) __builtin_amdgcn_s_wait_tensorcnt(0);
#endif
        __syncthreads();   // A stores + TDM tile visible to all waves

        // ---- fragments (16-bit A 16x32 / B 32x16 ISA layout) + 4 WMMAs ----
        FragBF af;
#pragma unroll
        for (int j = 0; j < 8; ++j) {
            const int ucol = (j < 4) ? (4 * h + j) : (4 + 4 * h + j);
            af.u[j] = As[wv * 16 + l16][ucol];
        }
#pragma unroll
        for (int s = 0; s < 4; ++s) {
            FragBF bf;
#pragma unroll
            for (int j = 0; j < 8; ++j) {
                const int ucol = (j < 4) ? (4 * h + j) : (4 + 4 * h + j);
                bf.u[j] = Bs[s * 16 + l16][ucol];
            }
            acc[s] = __builtin_amdgcn_wmma_f32_16x16x32_bf16(
                false, af.v, false, bf.v, (short)0, acc[s], false, false);
        }
    }

    // ---- epilogue: folded BN / bias, optional ReLU, NCHW scatter ----
    const int mbase = m0 + wv * 16;
#pragma unroll
    for (int s = 0; s < 4; ++s) {
        const int n = n0 + s * 16 + l16;
        float scale = 1.0f, shift = 0.0f;
        if (n < Cout) {
            if (mode == 2) {
                const float inv = p0[n] * rsqrtf(p3[n] + 1e-5f);
                scale = inv;
                shift = p1[n] - p2[n] * inv;
            } else {
                shift = p1[n];
            }
        }
#pragma unroll
        for (int i = 0; i < 8; ++i) {
            const int mo = mbase + i + h * 8;   // C layout: VGPR i -> M=i (+8 hi)
            if (mo < M && n < Cout) {
                float v = acc[s][i] * scale + shift;
                if (mode >= 1) v = fmaxf(v, 0.0f);
                const int b2 = mo / hw;
                const int rr = mo - b2 * hw;
                const int oy = rr / OW;
                const int ox = rr - oy * OW;
                y[(((size_t)b2 * Cout + n) * OH + oy) * OW + ox] = v;
            }
        }
    }
}

// ---------------------------------------------------------------------------
// GEMM-NT, all-bf16 inputs: out = relu?( A[M,K] * Wt[N,K]^T + bias )
// A staged as packed uint loads; W tile via TDM. Output f32 (strided, with
// column offset) or bf16 (obf=1, dense ld).
// ---------------------------------------------------------------------------
__global__ __launch_bounds__(256) void gemm_nt_wmma_kernel(
    const unsigned short* __restrict__ Abf, const unsigned short* __restrict__ Wbf,
    const float* __restrict__ bias, float* __restrict__ outf,
    unsigned short* __restrict__ outbf,
    int M, int N, int K, int out_ld, int col0, int relu, int obf)
{
    __shared__ unsigned int As[BM][LDW];
    __shared__ unsigned int Bs[BN][LDW];

    const int tid  = threadIdx.x;
    const int lane = tid & 31;
    const int wv   = tid >> 5;
    const int l16  = lane & 15;
    const int h    = lane >> 4;

    const int m0 = blockIdx.x * BM;
    const int n0 = blockIdx.y * BN;

    const unsigned int* arow[8];
#pragma unroll
    for (int j = 0; j < 8; ++j) {
        const int uidx = tid + 256 * j;
        const int row  = uidx >> 4;
        int mA = m0 + row; if (mA >= M) mA = M - 1;
        arow[j] = (const unsigned int*)(Abf + (size_t)mA * K);
    }

    v8f acc[4] = {v8f{}, v8f{}, v8f{}, v8f{}};

    for (int kb = 0; kb < K; kb += BK) {
        __syncthreads();
        if (wv == 0) {
            tdm_load_2d_bf16((unsigned int)(size_t)&Bs[0][0],
                             Wbf + (size_t)n0 * K + kb,
                             BK, BN, (unsigned long long)K,
                             (unsigned int)(K - kb), (unsigned int)(N - n0));
        }
#pragma unroll
        for (int j = 0; j < 8; ++j) {
            const int uidx = tid + 256 * j;
            const int row  = uidx >> 4;
            const int ucol = uidx & 15;
            const int ku   = (kb >> 1) + ucol;           // uint index (K even)
            if (kb + 2 * BK < K)
                __builtin_prefetch(arow[j] + ku + BK, 0, 0);  // global_prefetch_b8
            As[row][ucol] = (kb + ucol * 2 < K) ? arow[j][ku] : 0u;
        }
#if __has_builtin(__builtin_amdgcn_s_wait_tensorcnt)
        if (wv == 0) __builtin_amdgcn_s_wait_tensorcnt(0);
#endif
        __syncthreads();

        FragBF af;
#pragma unroll
        for (int j = 0; j < 8; ++j) {
            const int ucol = (j < 4) ? (4 * h + j) : (4 + 4 * h + j);
            af.u[j] = As[wv * 16 + l16][ucol];
        }
#pragma unroll
        for (int s = 0; s < 4; ++s) {
            FragBF bf;
#pragma unroll
            for (int j = 0; j < 8; ++j) {
                const int ucol = (j < 4) ? (4 * h + j) : (4 + 4 * h + j);
                bf.u[j] = Bs[s * 16 + l16][ucol];
            }
            acc[s] = __builtin_amdgcn_wmma_f32_16x16x32_bf16(
                false, af.v, false, bf.v, (short)0, acc[s], false, false);
        }
    }

    const int mbase = m0 + wv * 16;
#pragma unroll
    for (int s = 0; s < 4; ++s) {
        const int n = n0 + s * 16 + l16;
        const float bs = (n < N) ? bias[n] : 0.0f;
#pragma unroll
        for (int i = 0; i < 8; ++i) {
            const int mo = mbase + i + h * 8;
            if (mo < M && n < N) {
                float v = acc[s][i] + bs;
                if (relu) v = fmaxf(v, 0.0f);
                if (obf) outbf[(size_t)mo * out_ld + n] = f2bf(v);
                else     outf [(size_t)mo * out_ld + col0 + n] = v;
            }
        }
    }
}

// Anchor decode + sigmoid scores. cls: [B,9,26,26], reg: [B,36,26,26]
__global__ void decode_score_kernel(
    const float* __restrict__ cls, const float* __restrict__ reg,
    float* __restrict__ proposals, float* __restrict__ scores, int Bn)
{
    const int NA = 26 * 26 * 9;
    const int gid = blockIdx.x * blockDim.x + threadIdx.x;
    if (gid >= Bn * NA) return;
    const int b = gid / NA;
    const int i = gid - b * NA;
    const int a  = i % 9;
    const int t  = i / 9;
    const int xg = t % 26;
    const int yg = t / 26;

    const float s_tab[3] = {16.0f, 32.0f, 64.0f};
    const float r_tab[3] = {0.5f, 1.0f, 2.0f};
    const float s  = s_tab[a / 3];
    const float r  = r_tab[a % 3];
    const float sq = sqrtf(r);
    const float aw = s * sq;
    const float ah = s / sq;
    const float cxa = (xg + 0.5f) * 16.0f;
    const float cya = (yg + 0.5f) * 16.0f;

    const size_t hwoff = (size_t)yg * 26 + xg;
    const size_t rbase = (size_t)b * 36 * 676 + hwoff;
    const float d0 = reg[rbase + (size_t)(a * 4 + 0) * 676];
    const float d1 = reg[rbase + (size_t)(a * 4 + 1) * 676];
    const float d2 = reg[rbase + (size_t)(a * 4 + 2) * 676];
    const float d3 = reg[rbase + (size_t)(a * 4 + 3) * 676];

    const float pcx = d0 * aw + cxa;
    const float pcy = d1 * ah + cya;
    const float pw  = expf(fminf(d2, 4.0f)) * aw;
    const float ph  = expf(fminf(d3, 4.0f)) * ah;

    float bx1 = pcx - 0.5f * pw, by1 = pcy - 0.5f * ph;
    float bx2 = pcx + 0.5f * pw, by2 = pcy + 0.5f * ph;
    bx1 = fminf(fmaxf(bx1, 0.0f), 416.0f);
    by1 = fminf(fmaxf(by1, 0.0f), 416.0f);
    bx2 = fminf(fmaxf(bx2, 0.0f), 416.0f);
    by2 = fminf(fmaxf(by2, 0.0f), 416.0f);
    float* p = proposals + (size_t)gid * 4;
    p[0] = bx1; p[1] = by1; p[2] = bx2; p[3] = by2;

    const float logit = cls[((size_t)b * 9 + a) * 676 + hwoff];
    scores[gid] = 1.0f / (1.0f + expf(-logit));
}

// Per-batch top-100 selection (iterative max). One block per batch image.
__global__ void topk_kernel(
    const float* __restrict__ scores, const float* __restrict__ proposals,
    float* __restrict__ work, float* __restrict__ topb, int NA, int Ksel)
{
    const int b = blockIdx.x;
    const int tid = threadIdx.x;
    const float* sc = scores + (size_t)b * NA;
    float* wk = work + (size_t)b * NA;
    for (int i = tid; i < NA; i += blockDim.x) wk[i] = sc[i];
    __syncthreads();

    __shared__ float sval[256];
    __shared__ int   sidx[256];
    for (int t = 0; t < Ksel; ++t) {
        float best = -1e30f; int bi = 0;
        for (int i = tid; i < NA; i += blockDim.x) {
            const float v = wk[i];
            if (v > best) { best = v; bi = i; }
        }
        sval[tid] = best; sidx[tid] = bi;
        __syncthreads();
        for (int off = 128; off > 0; off >>= 1) {
            if (tid < off && sval[tid + off] > sval[tid]) {
                sval[tid] = sval[tid + off];
                sidx[tid] = sidx[tid + off];
            }
            __syncthreads();
        }
        if (tid == 0) {
            const int idx = sidx[0];
            const float* p = proposals + ((size_t)b * NA + idx) * 4;
            float* o = topb + ((size_t)b * Ksel + t) * 4;
            o[0] = p[0]; o[1] = p[1]; o[2] = p[2]; o[3] = p[3];
            wk[idx] = -1e30f;
        }
        __syncthreads();
    }
}

// ROI-align: feats [B,128,26,26] f32, boxes [B,100,4] -> pooled bf16 [1600,6272]
__global__ void roi_align_kernel(
    const float* __restrict__ feats, const float* __restrict__ topb,
    unsigned short* __restrict__ pooled, int Bn)
{
    const int total = Bn * 100 * 128 * 49;
    const int gid = blockIdx.x * blockDim.x + threadIdx.x;
    if (gid >= total) return;
    const int px = gid % 7;
    const int py = (gid / 7) % 7;
    const int c  = (gid / 49) % 128;
    const int k  = (gid / (49 * 128)) % 100;
    const int b  = gid / (49 * 128 * 100);

    const float* box = topb + ((size_t)b * 100 + k) * 4;
    const float scl = 1.0f / 16.0f;
    const float x1 = box[0] * scl;
    const float y1 = box[1] * scl;
    const float rw = fmaxf(box[2] * scl - x1, 1.0f);
    const float rh = fmaxf(box[3] * scl - y1, 1.0f);
    const float* f = feats + ((size_t)b * 128 + c) * 676;

    float sum = 0.0f;
#pragma unroll
    for (int sy = 0; sy < 2; ++sy) {
#pragma unroll
        for (int sx = 0; sx < 2; ++sx) {
            const float oy = ((py * 2 + sy) + 0.5f) / 14.0f;
            const float ox = ((px * 2 + sx) + 0.5f) / 14.0f;
            float Y = y1 + oy * rh;
            float X = x1 + ox * rw;
            Y = fminf(fmaxf(Y, 0.0f), 25.0f);
            X = fminf(fmaxf(X, 0.0f), 25.0f);
            const int y0  = (int)floorf(Y);
            const int x0  = (int)floorf(X);
            const int y1i = min(y0 + 1, 25);
            const int x1i = min(x0 + 1, 25);
            const float ly = Y - y0;
            const float lx = X - x0;
            const float v00 = f[y0 * 26 + x0];
            const float v01 = f[y0 * 26 + x1i];
            const float v10 = f[y1i * 26 + x0];
            const float v11 = f[y1i * 26 + x1i];
            sum += v00 * (1 - ly) * (1 - lx) + v01 * (1 - ly) * lx +
                   v10 * ly * (1 - lx)       + v11 * ly * lx;
        }
    }
    pooled[((size_t)b * 100 + k) * 6272 + (size_t)c * 49 + py * 7 + px] =
        f2bf(sum * 0.25f);
}

// ---------------------------------------------------------------------------
extern "C" void kernel_launch(void* const* d_in, const int* in_sizes, int n_in,
                              void* d_out, int out_size, void* d_ws, size_t ws_size,
                              hipStream_t stream)
{
    (void)in_sizes; (void)n_in; (void)out_size; (void)ws_size;
    const float* images = (const float*)d_in[0];
    const float* bw[4]; const float* bg[4]; const float* bb[4]; const float* bm[4]; const float* bv[4];
    for (int l = 0; l < 4; ++l) {
        bw[l] = (const float*)d_in[1 + l * 5 + 0];
        bg[l] = (const float*)d_in[1 + l * 5 + 1];
        bb[l] = (const float*)d_in[1 + l * 5 + 2];
        bm[l] = (const float*)d_in[1 + l * 5 + 3];
        bv[l] = (const float*)d_in[1 + l * 5 + 4];
    }
    const float* rpn_conv_w = (const float*)d_in[21];
    const float* rpn_conv_b = (const float*)d_in[22];
    const float* rpn_cls_w  = (const float*)d_in[23];
    const float* rpn_cls_b  = (const float*)d_in[24];
    const float* rpn_reg_w  = (const float*)d_in[25];
    const float* rpn_reg_b  = (const float*)d_in[26];
    const float* roi_w      = (const float*)d_in[27];
    const float* roi_b      = (const float*)d_in[28];
    const float* cls_w      = (const float*)d_in[29];
    const float* cls_b      = (const float*)d_in[30];
    const float* box_w      = (const float*)d_in[31];
    const float* box_b      = (const float*)d_in[32];
    float* out = (float*)d_out;

    // Workspace (~163 MB): ping at 0, pong at 96MB, persistents above 144MB.
    const size_t MB = 1024ull * 1024ull;
    char* ws = (char*)d_ws;
    float*          x1     = (float*)(ws + 0);          // 16x32x208x208 (88.6MB)
    float*          x2     = (float*)(ws + 96 * MB);    // 16x64x104x104 (44.3MB)
    float*          x3     = (float*)(ws + 0);          // 16x128x52x52, x1 dead
    float*          feats  = (float*)(ws + 144 * MB);   // 16x128x26x26 (5.3MB)
    float*          rbuf   = (float*)(ws + 96 * MB);    // 16x256x26x26, x2 dead
    float*          clsmap = (float*)(ws + 150 * MB);
    float*          regmap = (float*)(ws + 151 * MB);
    float*          props  = (float*)(ws + 153 * MB);
    float*          scores = (float*)(ws + 155 * MB);
    float*          work   = (float*)(ws + 156 * MB);
    float*          topb   = (float*)(ws + 157 * MB);
    unsigned short* pooled = (unsigned short*)(ws + 0); // 1600x6272 bf16, x3 dead
    unsigned short* hbuf   = (unsigned short*)(ws + 158 * MB);  // 1600x128 bf16

    // bf16 weight arena at 160MB (~2.6MB total)
    unsigned short* warena = (unsigned short*)(ws + 160 * MB);
    const int wsz[10] = {32 * 27, 64 * 288, 128 * 576, 128 * 1152,
                         256 * 1152, 9 * 256, 36 * 256, 128 * 6272,
                         2 * 128, 8 * 128};
    const float* wsrc[10] = {bw[0], bw[1], bw[2], bw[3], rpn_conv_w,
                             rpn_cls_w, rpn_reg_w, roi_w, cls_w, box_w};
    unsigned short* wdst[10];
    {
        size_t o = 0;
        for (int i = 0; i < 10; ++i) { wdst[i] = warena + o; o += (size_t)wsz[i]; }
    }
    for (int i = 0; i < 10; ++i)
        f2bf_kernel<<<dim3((wsz[i] + 255) / 256), dim3(256), 0, stream>>>(
            wsrc[i], wdst[i], wsz[i]);

    const dim3 blk(256);

    // Backbone conv+bn+relu (mode 2), stride 2, pad 1, 3x3.
    conv_wmma_kernel<<<dim3(5408, 1), blk, 0, stream>>>(    // 3->32, 416->208
        images, wdst[0], bg[0], bb[0], bm[0], bv[0], x1,
        16, 3, 416, 416, 32, 208, 208, 2, 1, 3, 2);
    conv_wmma_kernel<<<dim3(1352, 1), blk, 0, stream>>>(    // 32->64, 208->104
        x1, wdst[1], bg[1], bb[1], bm[1], bv[1], x2,
        16, 32, 208, 208, 64, 104, 104, 2, 1, 3, 2);
    conv_wmma_kernel<<<dim3(338, 2), blk, 0, stream>>>(     // 64->128, 104->52
        x2, wdst[2], bg[2], bb[2], bm[2], bv[2], x3,
        16, 64, 104, 104, 128, 52, 52, 2, 1, 3, 2);
    conv_wmma_kernel<<<dim3(85, 2), blk, 0, stream>>>(      // 128->128, 52->26
        x3, wdst[3], bg[3], bb[3], bm[3], bv[3], feats,
        16, 128, 52, 52, 128, 26, 26, 2, 1, 3, 2);

    // RPN conv 3x3 128->256, stride 1, pad 1, bias+relu (mode 1)
    conv_wmma_kernel<<<dim3(85, 4), blk, 0, stream>>>(
        feats, wdst[4], nullptr, rpn_conv_b, nullptr, nullptr, rbuf,
        16, 128, 26, 26, 256, 26, 26, 1, 1, 3, 1);
    // 1x1 heads, bias only (mode 0)
    conv_wmma_kernel<<<dim3(85, 1), blk, 0, stream>>>(
        rbuf, wdst[5], nullptr, rpn_cls_b, nullptr, nullptr, clsmap,
        16, 256, 26, 26, 9, 26, 26, 1, 0, 1, 0);
    conv_wmma_kernel<<<dim3(85, 1), blk, 0, stream>>>(
        rbuf, wdst[6], nullptr, rpn_reg_b, nullptr, nullptr, regmap,
        16, 256, 26, 26, 36, 26, 26, 1, 0, 1, 0);

    // Anchor decode + sigmoid
    decode_score_kernel<<<dim3((16 * 6084 + 255) / 256), dim3(256), 0, stream>>>(
        clsmap, regmap, props, scores, 16);

    // Top-100 per batch
    topk_kernel<<<dim3(16), dim3(256), 0, stream>>>(
        scores, props, work, topb, 6084, 100);

    // ROI align -> pooled bf16 [1600, 6272]
    roi_align_kernel<<<dim3((16 * 100 * 128 * 49 + 255) / 256), dim3(256), 0, stream>>>(
        feats, topb, pooled, 16);

    // ROI FC: [1600,6272] x [128,6272]^T + b, relu -> hbuf (bf16)
    gemm_nt_wmma_kernel<<<dim3(13, 2), blk, 0, stream>>>(
        pooled, wdst[7], roi_b, nullptr, hbuf, 1600, 128, 6272, 128, 0, 1, 1);

    // Heads into d_out [1600, 10]: cls (cols 0..1), box (cols 2..9)
    gemm_nt_wmma_kernel<<<dim3(13, 1), blk, 0, stream>>>(
        hbuf, wdst[8], cls_b, out, nullptr, 1600, 2, 128, 10, 0, 0, 0);
    gemm_nt_wmma_kernel<<<dim3(13, 1), blk, 0, stream>>>(
        hbuf, wdst[9], box_b, out, nullptr, 1600, 8, 128, 10, 2, 0, 0);
}
